// PointFeatureNet_91070486544465
// MI455X (gfx1250) — compile-verified, matrix operand order checked by reading, not run on hardware
//
#include <hip/hip_runtime.h>
#include <hip/hip_bf16.h>
#include <math.h>

typedef __attribute__((ext_vector_type(16))) _Float16 v16h;
typedef __attribute__((ext_vector_type(8)))  _Float16 v8h;
typedef __attribute__((ext_vector_type(8)))  float    v8f;
typedef __attribute__((ext_vector_type(4)))  float    v4f;
typedef __attribute__((ext_vector_type(4)))  unsigned int v4u;
typedef __attribute__((ext_vector_type(8)))  int      v8i;
typedef __attribute__((ext_vector_type(4)))  int      v4i;

constexpr int BB = 8, NN = 8192, NPT = 512, CUTK = 128, IN_CH = 64;
constexpr float EPS = 1e-5f;

static __device__ __forceinline__ v16h cat8(v8h lo, v8h hi) {
    return __builtin_shufflevector(lo, hi, 0,1,2,3,4,5,6,7,8,9,10,11,12,13,14,15);
}

// ---------------------------------------------------------------------------
// TDM: stage the whole f16 weight matrix (O x C, <= 24KB) into LDS with one
// TENSOR_LOAD_TO_LDS. D# packing per CDNA5 ISA #8:
//   group0: [1:0]=count=1, [63:32]=lds_addr, [120:64]=global_addr, [127:126]=2
//   group1: [17:16]=data_size=1 (2B), tensor_dim0=C, tensor_dim1=O,
//           tile_dim0=C, tile_dim1=O, tile_dim2=0, tensor_dim0_stride=C
//   groups 2/3: zero (2D tile)
// ---------------------------------------------------------------------------
static __device__ __forceinline__ void tdm_load_w(const _Float16* gw,
                                                  unsigned ldsOff, int O, int C)
{
    unsigned long long ga = (unsigned long long)gw;
    v4u g0;
    g0[0] = 1u;                                           // count = 1 (valid)
    g0[1] = ldsOff;                                       // lds_addr
    g0[2] = (unsigned)(ga & 0xffffffffu);                 // global_addr[31:0]
    g0[3] = (unsigned)((ga >> 32) & 0x01ffffffu)          // global_addr[56:32]
          | (2u << 30);                                   // type = 2 ("image")
    unsigned td0 = (unsigned)C, td1 = (unsigned)O;
    v8i g1;
    g1[0] = (int)(1u << 16);                              // data_size=1 (2B)
    g1[1] = (int)((td0 & 0xffffu) << 16);                 // tensor_dim0[15:0]
    g1[2] = (int)(((td0 >> 16) & 0xffffu) | ((td1 & 0xffffu) << 16));
    g1[3] = (int)(((td1 >> 16) & 0xffffu) | (td0 << 16)); // tile_dim0 = C
    g1[4] = (int)(td1 & 0xffffu);                         // tile_dim1=O, tile_dim2=0
    g1[5] = (int)td0;                                     // dim0_stride[31:0] = C
    g1[6] = 0;
    g1[7] = 0;
    v4i g2 = {0, 0, 0, 0};
    v4i g3 = {0, 0, 0, 0};
#if __clang_major__ >= 23
    v8i g4 = {0, 0, 0, 0, 0, 0, 0, 0};
    __builtin_amdgcn_tensor_load_to_lds(g0, g1, g2, g3, g4, 0);
#else
    __builtin_amdgcn_tensor_load_to_lds(g0, g1, g2, g3, 0);
#endif
}

// ---------------------------------------------------------------------------
// Farthest point sampling: one workgroup per batch, distance state in LDS.
// ---------------------------------------------------------------------------
__global__ __launch_bounds__(512)
void fps_kernel(const float* __restrict__ xyz, int* __restrict__ cents)
{
    const int b   = blockIdx.x;
    const int tid = threadIdx.x;
    __shared__ float dist[NN];
    __shared__ float rv[512];
    __shared__ int   ri[512];
    __shared__ int   cur;

    const float* px = xyz + (size_t)b * NN * 3;
    for (int i = tid; i < NN; i += 512) dist[i] = 1e10f;
    if (tid == 0) cur = (int)((2654435761u * (unsigned)(b + 1)) % (unsigned)NN);
    __syncthreads();

    for (int it = 0; it < NPT; ++it) {
        const int f = cur;
        if (tid == 0) cents[b * NPT + it] = f;
        const float cx = px[f*3+0], cy = px[f*3+1], cz = px[f*3+2];

        float bv = -1.0f; int bi = NN;
        for (int i = tid; i < NN; i += 512) {
            float dx = px[i*3+0]-cx, dy = px[i*3+1]-cy, dz = px[i*3+2]-cz;
            float d  = dx*dx + dy*dy + dz*dz;
            float nd = (d < dist[i]) ? truncf(d) : dist[i];
            dist[i] = nd;
            if (nd > bv) { bv = nd; bi = i; }   // ascending i => first max kept
        }
        rv[tid] = bv; ri[tid] = bi;
        __syncthreads();
        for (int st = 256; st > 0; st >>= 1) {
            if (tid < st) {
                if (rv[tid+st] > rv[tid] ||
                    (rv[tid+st] == rv[tid] && ri[tid+st] < ri[tid])) {
                    rv[tid] = rv[tid+st]; ri[tid] = ri[tid+st];
                }
            }
            __syncthreads();
        }
        if (tid == 0) cur = ri[0];
        __syncthreads();
    }
}

// ---------------------------------------------------------------------------
// Per-centroid 128-NN selection (sorted ascending), one workgroup per (b,s).
// ---------------------------------------------------------------------------
__global__ __launch_bounds__(256)
void knn_topk_kernel(const float* __restrict__ xyz,
                     const int* __restrict__ cents, int* __restrict__ nbr)
{
    const int bs  = blockIdx.x;
    const int b   = bs / NPT;
    const int tid = threadIdx.x;
    __shared__ float d[NN];
    __shared__ float rv[256];
    __shared__ int   ri[256];

    const float* px = xyz + (size_t)b * NN * 3;
    const int c = cents[bs];
    const float cx = px[c*3+0], cy = px[c*3+1], cz = px[c*3+2];

    for (int i = tid; i < NN; i += 256) {
        float dx = px[i*3+0]-cx, dy = px[i*3+1]-cy, dz = px[i*3+2]-cz;
        d[i] = dx*dx + dy*dy + dz*dz;
    }
    __syncthreads();

    float bv = 1e30f; int bi = NN;
    for (int i = tid; i < NN; i += 256)
        if (d[i] < bv) { bv = d[i]; bi = i; }

    for (int j = 0; j < CUTK; ++j) {
        rv[tid] = bv; ri[tid] = bi;
        __syncthreads();
        for (int st = 128; st > 0; st >>= 1) {
            if (tid < st) {
                if (rv[tid+st] < rv[tid] ||
                    (rv[tid+st] == rv[tid] && ri[tid+st] < ri[tid])) {
                    rv[tid] = rv[tid+st]; ri[tid] = ri[tid+st];
                }
            }
            __syncthreads();
        }
        const int w = ri[0];
        if (tid == 0) nbr[(size_t)bs * CUTK + j] = w;
        __syncthreads();
        if ((w & 255) == tid) {                  // owner invalidates + rescans
            d[w] = 1e30f;
            bv = 1e30f; bi = NN;
            for (int i = tid; i < NN; i += 256)
                if (d[i] < bv) { bv = d[i]; bi = i; }
        }
        __syncthreads();
    }
}

__global__ void xyz_new_kernel(const float* __restrict__ xyz,
                               const int* __restrict__ cents,
                               float* __restrict__ out)
{
    int i = blockIdx.x * blockDim.x + threadIdx.x;
    if (i >= BB * NPT) return;
    int b = i / NPT;
    int c = cents[i];
    const float* p = xyz + ((size_t)b * NN + c) * 3;
    out[i*3+0] = p[0]; out[i*3+1] = p[1]; out[i*3+2] = p[2];
}

// ---------------------------------------------------------------------------
// Gather concat(features, xyz) -> f16, zero-padded to Cpad=96; fully
// vectorized: b128 loads of features, b128 (v8h) stores of the f16 row.
// ---------------------------------------------------------------------------
__global__ void gather_kernel(const float* __restrict__ xyz,
                              const float* __restrict__ feat,
                              const int* __restrict__ nbr,
                              _Float16* __restrict__ X, int K, int Cpad, int M)
{
    int m = blockIdx.x * blockDim.x + threadIdx.x;
    if (m >= M) return;
    int k = m % K;
    int s = (m / K) % NPT;
    int b = m / (K * NPT);
    int idx = nbr[((size_t)(b * NPT + s)) * CUTK + k];
    const v4f* f4 = (const v4f*)(feat + ((size_t)b * NN + idx) * IN_CH);
    const float* p = xyz + ((size_t)b * NN + idx) * 3;
    _Float16* xr = X + (size_t)m * Cpad;
    #pragma unroll
    for (int c8 = 0; c8 < 8; ++c8) {             // channels 0..63
        v4f a = f4[c8*2], bq = f4[c8*2+1];
        v8h h;
        #pragma unroll
        for (int j = 0; j < 4; ++j) { h[j] = (_Float16)a[j]; h[4+j] = (_Float16)bq[j]; }
        *(v8h*)(xr + c8*8) = h;
    }
    v8h t = {};
    t[0] = (_Float16)p[0]; t[1] = (_Float16)p[1]; t[2] = (_Float16)p[2];
    *(v8h*)(xr + 64) = t;                        // 64..66 xyz, 67..71 zero
    v8h z = {};
    *(v8h*)(xr + 72) = z; *(v8h*)(xr + 80) = z; *(v8h*)(xr + 88) = z;
}

__global__ void wconv_kernel(const float* __restrict__ W,
                             _Float16* __restrict__ Wh, int O, int Cin, int Cpad)
{
    int i = blockIdx.x * blockDim.x + threadIdx.x;
    if (i >= O * Cpad) return;
    int o = i / Cpad, c = i % Cpad;
    Wh[i] = (c < Cin) ? (_Float16)W[o * Cin + c] : (_Float16)0.f;
}

__global__ void zero_kernel(float* p, int n)
{
    int i = blockIdx.x * blockDim.x + threadIdx.x;
    if (i < n) p[i] = 0.f;
}

// ---------------------------------------------------------------------------
// WMMA GEMM: Y[m][o] = sum_c X[m][c] * Wh[o][c]  (f16 in, f32 acc, f16 out)
// Weights staged into LDS once per block via TDM (wave 0 issues
// tensor_load_to_lds, s_wait_tensorcnt 0, then workgroup barrier); B operand
// fragments then come from LDS (ds_load_b128), A streamed from global.
// Per-channel BN statistics: shfl_xor(16) lane-half fold, LDS ds_add_f32,
// one global atomic per channel per block.
// ---------------------------------------------------------------------------
__global__ __launch_bounds__(256)
void wmma_gemm_stats(const _Float16* __restrict__ X,
                     const _Float16* __restrict__ Wh,
                     _Float16* __restrict__ Y,
                     float* __restrict__ stats,   // [0..O): sum, [O..2O): sumsq
                     int M, int C, int O)
{
    extern __shared__ _Float16 lds_w[];           // O*C f16 weights
    const int tid  = threadIdx.x;
    const int wave = tid >> 5;
    const int lane = tid & 31;
    const int half = lane >> 4;     // lane-half selects K sub-blocks
    const int lm   = lane & 15;     // row (A) / col (B) within the tile
    const int m0   = blockIdx.x * 128 + wave * 16;
    const int n0   = blockIdx.y * 16;

    if (tid < 32) {                 // wave 0 drives the TDM (EXEC-independent)
        tdm_load_w(Wh, (unsigned)(unsigned long long)(void*)lds_w, O, C);
        __builtin_amdgcn_s_wait_tensorcnt(0);
    }
    __syncthreads();

    const _Float16* xrow = X + (size_t)(m0 + lm) * C + half * 8;
    const _Float16* wrow = lds_w + (n0 + lm) * C + half * 8;

    v8f acc = {};
    for (int kc = 0; kc < C; kc += 32) {
        // A 16x32 f16: lanes 0-15 hold K {0..7,16..23}; lanes 16-31 {8..15,24..31}
        v8h alo = *(const v8h*)(xrow + kc);
        v8h ahi = *(const v8h*)(xrow + kc + 16);
        // B 32x16 f16 from LDS: mirrored layout, column per lane, B[k][n]=W[n][k]
        v8h blo = *(const v8h*)(wrow + kc);
        v8h bhi = *(const v8h*)(wrow + kc + 16);
        if (kc + 32 < C) {
            __builtin_prefetch(xrow + kc + 32, 0, 3);   // global_prefetch_b8
        }
        v16h a = cat8(alo, ahi);
        v16h b = cat8(blo, bhi);
        acc = __builtin_amdgcn_wmma_f32_16x16x32_f16(
                  false, a, false, b, (short)0, acc, false, false);
    }

    // D layout: VGPR r -> M = m0 + r (lanes 0-15) / m0 + 8 + r (lanes 16-31),
    // N = n0 + (lane & 15).
    float s = 0.f, sq = 0.f;
    #pragma unroll
    for (int r = 0; r < 8; ++r) {
        float v = acc[r];
        Y[(size_t)(m0 + r + half * 8) * O + (n0 + lm)] = (_Float16)v;
        s += v; sq += v * v;
    }
    s  += __shfl_xor(s, 16, 32);
    sq += __shfl_xor(sq, 16, 32);

    __shared__ float lsum[16], lsq[16];
    if (tid < 16) { lsum[tid] = 0.f; lsq[tid] = 0.f; }
    __syncthreads();
    if (half == 0) {                         // ds_add_f32 per channel per wave
        atomicAdd(&lsum[lm], s);
        atomicAdd(&lsq[lm], sq);
    }
    __syncthreads();
    if (tid < 16) {
        atomicAdd(&stats[n0 + tid],     lsum[tid]);
        atomicAdd(&stats[O + n0 + tid], lsq[tid]);
    }
}

// ---------------------------------------------------------------------------
// BatchNorm + ReLU -> next layer's f16 input, 8 f16 per thread (b128 I/O).
// Linear bias cancels under BN and is omitted throughout.
// ---------------------------------------------------------------------------
__global__ void bn_relu_kernel(const _Float16* __restrict__ Y,
                               const float* __restrict__ stats,
                               const float* __restrict__ gamma,
                               const float* __restrict__ beta,
                               _Float16* __restrict__ Xn, int M, int O)
{
    long t = (long)blockIdx.x * blockDim.x + threadIdx.x;
    long base = t * 8;
    if (base >= (long)M * O) return;
    int o0 = (int)(base % O);                    // O % 8 == 0: chunk stays in-row
    float invM = 1.0f / (float)M;
    v8h y = *(const v8h*)(Y + base);
    v8h r;
    #pragma unroll
    for (int j = 0; j < 8; ++j) {
        int o = o0 + j;
        float mean = stats[o] * invM;
        float var  = stats[O + o] * invM - mean * mean;
        float sc   = gamma[o] * rsqrtf(var + EPS);
        float sh   = beta[o] - mean * sc;
        float v    = (float)y[j] * sc + sh;
        r[j] = (_Float16)fmaxf(v, 0.f);
    }
    *(v8h*)(Xn + base) = r;
}

// Final layer: BN + ReLU + max over the K neighbors -> f32 output slice.
__global__ __launch_bounds__(128)
void bn_relu_max_kernel(const _Float16* __restrict__ Y,
                        const float* __restrict__ stats,
                        const float* __restrict__ gamma,
                        const float* __restrict__ beta,
                        float* __restrict__ out, int K, int branch_off, int M)
{
    const int bs = blockIdx.x;        // b*NPT + s
    const int o  = threadIdx.x;       // O == 128
    float invM = 1.0f / (float)M;
    float mean = stats[o] * invM;
    float var  = stats[128 + o] * invM - mean * mean;
    float sc   = gamma[o] * rsqrtf(var + EPS);
    float sh   = beta[o] - mean * sc;
    const _Float16* y = Y + (size_t)bs * K * 128;
    float mx = -1e30f;
    for (int k = 0; k < K; ++k) {
        float v = (float)y[(size_t)k * 128 + o] * sc + sh;
        v = fmaxf(v, 0.f);
        mx = fmaxf(mx, v);
    }
    out[(size_t)BB * NPT * 3 + (size_t)bs * 384 + branch_off + o] = mx;
}

// ---------------------------------------------------------------------------
extern "C" void kernel_launch(void* const* d_in, const int* in_sizes, int n_in,
                              void* d_out, int out_size, void* d_ws, size_t ws_size,
                              hipStream_t stream)
{
    const float* xyz  = (const float*)d_in[0];
    const float* feat = (const float*)d_in[1];

    const float *W[3][3], *gamma[3][3], *beta[3][3];
    int li = 2;
    for (int i = 0; i < 3; ++i)
        for (int j = 0; j < 3; ++j) {
            W[i][j]     = (const float*)d_in[li + 0];
            // d_in[li+1] is the linear bias: cancels under BatchNorm, unused.
            gamma[i][j] = (const float*)d_in[li + 2];
            beta[i][j]  = (const float*)d_in[li + 3];
            li += 4;
        }

    static const int KER[3]     = {16, 32, 128};
    static const int OUTS[3][3] = {{64, 64, 128}, {64, 96, 128}, {64, 96, 128}};
    static const int CINS[3][3] = {{67, 64, 64},  {67, 64, 96},  {67, 64, 96}};
    static const int CPAD[3][3] = {{96, 64, 64},  {96, 64, 96},  {96, 64, 96}};

    char* base = (char*)d_ws;
    size_t off = 0;
    auto alloc = [&](size_t bytes) -> void* {
        void* p = base + off;
        off = (off + bytes + 255) & ~(size_t)255;
        return p;
    };
    int*   cents = (int*)alloc((size_t)BB * NPT * 4);
    int*   nbr   = (int*)alloc((size_t)BB * NPT * CUTK * 4);
    float* stats = (float*)alloc(256 * 4);
    _Float16* Wh[3][3];
    for (int i = 0; i < 3; ++i)
        for (int j = 0; j < 3; ++j)
            Wh[i][j] = (_Float16*)alloc((size_t)128 * 96 * 2);
    _Float16* bufA = (_Float16*)alloc((size_t)BB * NPT * 128 * 96  * 2); //  96 MB
    _Float16* bufB = (_Float16*)alloc((size_t)BB * NPT * 128 * 128 * 2); // 128 MB
    float* out = (float*)d_out;

    for (int i = 0; i < 3; ++i)
        for (int j = 0; j < 3; ++j) {
            int tot = OUTS[i][j] * CPAD[i][j];
            wconv_kernel<<<(tot + 255) / 256, 256, 0, stream>>>(
                W[i][j], Wh[i][j], OUTS[i][j], CINS[i][j], CPAD[i][j]);
        }

    fps_kernel<<<BB, 512, 0, stream>>>(xyz, cents);
    knn_topk_kernel<<<BB * NPT, 256, 0, stream>>>(xyz, cents, nbr);
    xyz_new_kernel<<<(BB * NPT + 255) / 256, 256, 0, stream>>>(xyz, cents, out);

    for (int i = 0; i < 3; ++i) {
        const int K = KER[i];
        const int M = BB * NPT * K;
        gather_kernel<<<M / 256, 256, 0, stream>>>(xyz, feat, nbr, bufA, K,
                                                   CPAD[i][0], M);
        for (int j = 0; j < 3; ++j) {
            const int C = CPAD[i][j];
            const int O = OUTS[i][j];
            zero_kernel<<<1, 256, 0, stream>>>(stats, 256);
            dim3 grid(M / 128, O / 16);
            size_t ldsW = (size_t)O * C * sizeof(_Float16);
            wmma_gemm_stats<<<grid, 256, ldsW, stream>>>(bufA, Wh[i][j], bufB,
                                                         stats, M, C, O);
            if (j < 2) {
                long tot = (long)M * O;
                bn_relu_kernel<<<(unsigned)(tot / 8 / 256), 256, 0, stream>>>(
                    bufB, stats, gamma[i][j], beta[i][j], bufA, M, O);
            } else {
                bn_relu_max_kernel<<<BB * NPT, 128, 0, stream>>>(
                    bufB, stats, gamma[i][j], beta[i][j], out, K, i * 128, M);
            }
        }
    }
}
